// LstmReg_5540507812169
// MI455X (gfx1250) — matrix operand order repaired and checked by reading, not compiled, not run on hardware
//
#include <hip/hip_runtime.h>
#include <hip/hip_bf16.h>
#include <stdint.h>

#define H    2048
#define B    64
#define T    1024
#define F    256
#define NG   (4 * H)        // 8192 gate columns
#define NWG  (H / 16)       // 128 persistent workgroups
#define NT_TILES (NG / 16)  // 512 N-tiles in swizzled weights
#define CW   128            // K-chunk width staged per barrier
#define NCHUNK ((F + H) / CW)  // 18 chunks: 2 for x@Wk, 16 for h@Wr
#define SA_STR 136          // 128 + 8 pad: 272B row stride (16B aligned, DS-conflict free)
#define KT_STRIDE_Q (NT_TILES * 512 / 8)  // u32x4 stride between K-tiles = 32768 (524288 B)

typedef __attribute__((ext_vector_type(16))) __bf16       v16bf;
typedef __attribute__((ext_vector_type(8)))  float        v8f;
typedef __attribute__((ext_vector_type(4)))  unsigned int u32x4;
typedef __attribute__((ext_vector_type(4)))  int          i32x4;

union AFrag { u32x4 q[2]; v16bf v; };

// ---- CDNA5 async global->LDS path (guarded; VGPR-bounce fallback) ----
#if defined(__HIP_DEVICE_COMPILE__) && __has_builtin(__builtin_amdgcn_global_load_async_to_lds_b128)
#define USE_ASYNC_LDS 1
#else
#define USE_ASYNC_LDS 0
#endif

#if USE_ASYNC_LDS
#if __has_builtin(__builtin_amdgcn_s_wait_asynccnt)
#define WAIT_ASYNC() __builtin_amdgcn_s_wait_asynccnt(0)
#else
#define WAIT_ASYNC() asm volatile("s_wait_asynccnt 0" ::: "memory")
#endif
#define GAS __attribute__((address_space(1)))
#define LAS __attribute__((address_space(3)))
#else
#define WAIT_ASYNC()
#endif

__device__ __forceinline__ unsigned short f2bf(float f) {
  unsigned u = __float_as_uint(f);
  u += 0x7FFFu + ((u >> 16) & 1u);          // round-to-nearest-even
  return (unsigned short)(u >> 16);
}
__device__ __forceinline__ float bf2f(unsigned short s) {
  return __uint_as_float(((unsigned)s) << 16);
}

// ---------------- prep kernels ----------------
__global__ void k_cvt_bf16(const float* __restrict__ src,
                           unsigned short* __restrict__ dst, int n) {
  int i = blockIdx.x * blockDim.x + threadIdx.x;
  if (i < n) dst[i] = f2bf(src[i]);
}

// w: [kdim, NG] row-major fp32 -> bf16 tiles [kdim/32][NG/16][lane 0..31][16]
// B-fragment layout: lanes 0-15 hold K=k0+e, lanes 16-31 hold K=k0+16+e; N=n0+(lane&15).
__global__ void k_swizzle_w(const float* __restrict__ w,
                            unsigned short* __restrict__ dst, int kdim) {
  int i = blockIdx.x * blockDim.x + threadIdx.x;
  if (i >= kdim * NG) return;
  int tile   = i >> 9;
  int within = i & 511;
  int lane   = within >> 4;
  int e      = within & 15;
  int kt = tile / NT_TILES;
  int nt = tile - kt * NT_TILES;
  int k  = kt * 32 + (lane >> 4) * 16 + e;
  int n  = nt * 16 + (lane & 15);
  dst[i] = f2bf(w[(size_t)k * NG + n]);
}

__global__ void k_zero_barrier(unsigned* p) { p[threadIdx.x] = 0u; }

// ---------------- device-wide step barrier ----------------
__device__ __forceinline__ void grid_sync(unsigned* cnt, unsigned* gen) {
  __threadfence();
  __syncthreads();
  if (threadIdx.x == 0) {
    unsigned g   = __hip_atomic_load(gen, __ATOMIC_RELAXED, __HIP_MEMORY_SCOPE_AGENT);
    unsigned old = __hip_atomic_fetch_add(cnt, 1u, __ATOMIC_ACQ_REL, __HIP_MEMORY_SCOPE_AGENT);
    if (old == NWG - 1) {
      __hip_atomic_store(cnt, 0u, __ATOMIC_RELAXED, __HIP_MEMORY_SCOPE_AGENT);
      __hip_atomic_store(gen, g + 1u, __ATOMIC_RELEASE, __HIP_MEMORY_SCOPE_AGENT);
    } else {
      while (__hip_atomic_load(gen, __ATOMIC_ACQUIRE, __HIP_MEMORY_SCOPE_AGENT) == g)
        __builtin_amdgcn_s_sleep(2);
    }
  }
  __syncthreads();
  __threadfence();
}

// per-thread source pointer for staging chunk c (row sr, 64-elem half sc)
__device__ __forceinline__ const unsigned short*
chunk_src(int c, int t, const unsigned short* __restrict__ x_bf,
          const unsigned short* __restrict__ hin, int sr, int sc) {
  if (c < F / CW) return x_bf + (((size_t)sr * T + t) * F + c * CW + sc);
  return hin + ((size_t)sr * H + (c * CW - F) + sc);
}

// ---------------- persistent LSTM kernel ----------------
// 128 WGs x 128 threads (4 waves). WG owns h columns [wg*16, wg*16+16).
// Wave w computes gate w (i,f,g,o) for all 64 batch rows (4 M-tiles of 16).
__global__ __launch_bounds__(128, 1)
void k_lstm(const unsigned short* __restrict__ x_bf,   // [B,T,F] bf16
            const unsigned short* __restrict__ wk_sw,  // swizzled [F/32][512][32][16]
            const unsigned short* __restrict__ wr_sw,  // swizzled [H/32][512][32][16]
            const float* __restrict__ bias,            // [4H]
            unsigned short* __restrict__ h0buf,        // [B*H] bf16 (double buffer 0)
            unsigned short* __restrict__ h1buf,        // [B*H] bf16 (double buffer 1)
            unsigned* bar_cnt, unsigned* bar_gen)
{
  const int wg   = blockIdx.x;
  const int tid  = threadIdx.x;
  const int wave = tid >> 5;      // gate id: 0=i,1=f,2=g,3=o
  const int lane = tid & 31;
  const int n0   = wg * 16;

  __shared__ unsigned short sA[2][64 * SA_STR];  // ping-pong 64xCW A-tile stage
  __shared__ float sZ[4][64 * 16];               // per-gate pre-activation tiles
  __shared__ float sC[64 * 16];                  // resident cell state slice

  // h(0) = 0, c(0) = 0
  for (int i = tid; i < 64 * 16; i += 128) sC[i] = 0.0f;
  for (int i = tid; i < 64 * 16; i += 128) {
    int r = i >> 4, c = i & 15;
    h0buf[(size_t)r * H + n0 + c] = 0;
  }
  grid_sync(bar_cnt, bar_gen);

  const float bias_n = bias[wave * H + n0 + (lane & 15)];  // col-broadcast matches C/D layout

  const int sr = tid >> 1;         // staging row 0..63
  const int sc = (tid & 1) * 64;   // staging 64-element half (128B per thread)
  const int kb = (lane >> 4) * 16; // A-fragment K half: lanes 0-15 K=0..15, lanes 16-31 K=16..31
  const int ml = lane & 15;

  // uniform tile-column offset inside swizzled weights (scalar per wave)
  const size_t wcol = (size_t)(wave * NWG + wg) * 512;

  for (int t = 0; t < T; ++t) {
    const unsigned short* __restrict__ hin  = (t & 1) ? h1buf : h0buf;
    unsigned short* __restrict__       hout = (t & 1) ? h0buf : h1buf;

    v8f acc[4];
    for (int mt = 0; mt < 4; ++mt)
      for (int e = 0; e < 8; ++e) acc[mt][e] = bias_n;

    // ---- prologue: stage chunk 0 ----
#if USE_ASYNC_LDS
    {
      const unsigned short* src = chunk_src(0, t, x_bf, hin, sr, sc);
      unsigned short* dst = &sA[0][sr * SA_STR + sc];
#pragma unroll
      for (int i = 0; i < 8; ++i)
        __builtin_amdgcn_global_load_async_to_lds_b128(
            (GAS i32x4*)(src + i * 8), (LAS i32x4*)(dst + i * 8), 0, 0);
    }
#else
    u32x4 pf[8];
    {
      const u32x4* s = (const u32x4*)chunk_src(0, t, x_bf, hin, sr, sc);
#pragma unroll
      for (int i = 0; i < 8; ++i) pf[i] = s[i];
      u32x4* d = (u32x4*)&sA[0][sr * SA_STR + sc];
#pragma unroll
      for (int i = 0; i < 8; ++i) d[i] = pf[i];
    }
#endif

    // ---- pipelined K loop: 18 chunks, one barrier each ----
    for (int c = 0; c < NCHUNK; ++c) {
      WAIT_ASYNC();          // this wave's chunk-c async loads complete
      __syncthreads();       // all waves: buf[c&1] ready, buf[c^1] no longer read

#if USE_ASYNC_LDS
      if (c + 1 < NCHUNK) {  // overlap chunk c+1 loads with chunk c WMMAs
        const unsigned short* src = chunk_src(c + 1, t, x_bf, hin, sr, sc);
        unsigned short* dst = &sA[(c + 1) & 1][sr * SA_STR + sc];
#pragma unroll
        for (int i = 0; i < 8; ++i)
          __builtin_amdgcn_global_load_async_to_lds_b128(
              (GAS i32x4*)(src + i * 8), (LAS i32x4*)(dst + i * 8), 0, 0);
      }
#else
      if (c + 1 < NCHUNK) {
        const u32x4* s = (const u32x4*)chunk_src(c + 1, t, x_bf, hin, sr, sc);
#pragma unroll
        for (int i = 0; i < 8; ++i) pf[i] = s[i];
      }
#endif

      // ---- compute chunk c: 4 K-slices x 4 M-tiles = 16 WMMAs ----
      const unsigned short* sbuf = sA[c & 1];
      // one per-chunk B base; the 4 K-tiles sit at constant 524288B strides
      // (fits the 24-bit signed instruction offset -> single-clause imm loads)
      const u32x4* bp;
      {
        const unsigned short* wsrc = (c < F / CW) ? wk_sw : wr_sw;
        const int kt0 = (c < F / CW) ? c * 4 : (c - F / CW) * 4;
        bp = (const u32x4*)(wsrc + (size_t)kt0 * (NT_TILES * 512) + wcol + lane * 16);
      }
#pragma unroll
      for (int ks = 0; ks < 4; ++ks) {
        AFrag bf;
        bf.q[0] = bp[ks * KT_STRIDE_Q];
        bf.q[1] = bp[ks * KT_STRIDE_Q + 1];
        AFrag af[4];
#pragma unroll
        for (int mt = 0; mt < 4; ++mt) {
          af[mt].q[0] = *(const u32x4*)&sbuf[(mt * 16 + ml) * SA_STR + ks * 32 + kb];
          af[mt].q[1] = *(const u32x4*)&sbuf[(mt * 16 + ml) * SA_STR + ks * 32 + kb + 8];
        }
#pragma unroll
        for (int mt = 0; mt < 4; ++mt)
          acc[mt] = __builtin_amdgcn_wmma_f32_16x16x32_bf16(
              false, af[mt].v, false, bf.v, (short)0, acc[mt], false, false);
      }

#if !USE_ASYNC_LDS
      if (c + 1 < NCHUNK) {
        u32x4* d = (u32x4*)&sA[(c + 1) & 1][sr * SA_STR + sc];
#pragma unroll
        for (int i = 0; i < 8; ++i) d[i] = pf[i];
      }
#endif
    }

    // ---- gates: scatter C/D tiles to LDS, elementwise LSTM update ----
    {
      const int hl = lane >> 4;
#pragma unroll
      for (int mt = 0; mt < 4; ++mt)
#pragma unroll
        for (int v = 0; v < 8; ++v)
          sZ[wave][(mt * 16 + hl * 8 + v) * 16 + ml] = acc[mt][v];
    }
    __syncthreads();
    for (int i = tid; i < 64 * 16; i += 128) {
      float ig = 1.0f / (1.0f + __expf(-sZ[0][i]));
      float fg = 1.0f / (1.0f + __expf(-sZ[1][i]));
      float gg = tanhf(sZ[2][i]);
      float og = 1.0f / (1.0f + __expf(-sZ[3][i]));
      float c  = fg * sC[i] + ig * gg;
      sC[i] = c;
      float h = og * tanhf(c);
      int r = i >> 4, cc = i & 15;
      hout[(size_t)r * H + n0 + cc] = f2bf(h);
    }
    grid_sync(bar_cnt, bar_gen);
  }
}

// ---------------- tiny head ----------------
__global__ void k_head1(const unsigned short* __restrict__ hbf,
                        const float* __restrict__ Wd, const float* __restrict__ bd,
                        float* __restrict__ y) {
  int i = blockIdx.x * blockDim.x + threadIdx.x;
  if (i >= B * 512) return;
  int b = i >> 9, j = i & 511;
  float s = bd[j];
  const unsigned short* hr = hbf + (size_t)b * H;
  for (int k = 0; k < H; ++k) s += bf2f(hr[k]) * Wd[(size_t)k * 512 + j];
  y[i] = fmaxf(s, 0.0f);
}

__global__ void k_head2(const float* __restrict__ y, const float* __restrict__ Wo,
                        const float* __restrict__ bo, float* __restrict__ out) {
  int b = threadIdx.x;
  if (b >= B) return;
  float s = bo[0];
  for (int j = 0; j < 512; ++j) s += y[b * 512 + j] * Wo[j];
  out[b] = fmaxf(s, 0.0f);
}

// ---------------- host ----------------
extern "C" void kernel_launch(void* const* d_in, const int* in_sizes, int n_in,
                              void* d_out, int out_size, void* d_ws, size_t ws_size,
                              hipStream_t stream) {
  (void)in_sizes; (void)n_in; (void)out_size; (void)ws_size;
  const float* x  = (const float*)d_in[0];
  const float* Wk = (const float*)d_in[1];
  const float* Wr = (const float*)d_in[2];
  const float* bb = (const float*)d_in[3];
  const float* Wd = (const float*)d_in[4];
  const float* bd = (const float*)d_in[5];
  const float* Wo = (const float*)d_in[6];
  const float* bo = (const float*)d_in[7];
  float* out = (float*)d_out;

  char* ws = (char*)d_ws;
  size_t off = 0;
  auto alloc = [&](size_t bytes) -> char* {
    char* p = ws + off;
    off = (off + bytes + 255) & ~(size_t)255;
    return p;
  };
  unsigned short* x_bf  = (unsigned short*)alloc((size_t)B * T * F * 2);  // 33.6 MB
  unsigned short* wk_sw = (unsigned short*)alloc((size_t)F * NG * 2);     //  4.2 MB
  unsigned short* wr_sw = (unsigned short*)alloc((size_t)H * NG * 2);     // 33.6 MB
  unsigned short* h0    = (unsigned short*)alloc((size_t)B * H * 2);
  unsigned short* h1    = (unsigned short*)alloc((size_t)B * H * 2);
  float*          ybuf  = (float*)alloc((size_t)B * 512 * 4);
  unsigned*       bar   = (unsigned*)alloc(256);

  k_zero_barrier<<<1, 2, 0, stream>>>(bar);
  { int n = B * T * F; k_cvt_bf16<<<(n + 255) / 256, 256, 0, stream>>>(x, x_bf, n); }
  { int n = F * NG;    k_swizzle_w<<<(n + 255) / 256, 256, 0, stream>>>(Wk, wk_sw, F); }
  { int n = H * NG;    k_swizzle_w<<<(n + 255) / 256, 256, 0, stream>>>(Wr, wr_sw, H); }
  k_lstm<<<NWG, 128, 0, stream>>>(x_bf, wk_sw, wr_sw, bb, h0, h1, bar, bar + 1);
  // T = 1024 (even): final h lives in h0
  { int n = B * 512;   k_head1<<<(n + 255) / 256, 256, 0, stream>>>(h0, Wd, bd, ybuf); }
  k_head2<<<1, 64, 0, stream>>>(ybuf, Wo, bo, out);
}
